// AXSConv2d_74852690034810
// MI455X (gfx1250) — matrix operand order repaired and checked by reading, not compiled, not run on hardware
//
#include <hip/hip_runtime.h>

typedef __attribute__((ext_vector_type(16))) __bf16 v16bf;
typedef __attribute__((ext_vector_type(8)))  float  v8f;
typedef int v4i_t __attribute__((vector_size(16)));
typedef __attribute__((address_space(1))) v4i_t* gptr128;
typedef __attribute__((address_space(3))) v4i_t* lptr128;

#define CIN   256
#define COUT  512
#define HH    56
#define WWID  56
#define HP    58            // padded H/W (1-wide zero halo)
#define HWX   3136          // 56*56
#define NBAT  32
#define MTOT  (NBAT * HWX)  // 100352
#define KTAP  2304          // CIN*9
#define BM    128
#define BN    128
#define AST   40            // LDS row stride in bf16 elems (80B: 16B-aligned, bank-staggered)
#define ABUF  (BM * AST)    // 5120 bf16 per tile buffer
#define NSTG  72            // 9 taps * 8 k-chunks of 32
#define WPACK_ELEMS (9 * COUT * CIN)
#define XPAD_ELEMS  (NBAT * HP * HP * CIN)   // 27,557,888 bf16

#if __has_builtin(__builtin_amdgcn_global_load_async_to_lds_b128) && \
    __has_builtin(__builtin_amdgcn_s_wait_asynccnt)
#define USE_ASYNC 1
#else
#define USE_ASYNC 0
#endif

__device__ __forceinline__ unsigned short f2bf(float f) {
  unsigned u = __float_as_uint(f);
  u += 0x7FFFu + ((u >> 16) & 1u);   // round-to-nearest-even truncation
  return (unsigned short)(u >> 16);
}

__device__ __forceinline__ void cp16_g2l(const unsigned short* g, unsigned short* l) {
#if USE_ASYNC
  __builtin_amdgcn_global_load_async_to_lds_b128(
      (gptr128)(uintptr_t)g,   // AS1: flat and global share representation
      (lptr128)(uintptr_t)l,   // AS3: flat LDS addr truncates to 32-bit LDS offset
      0, 0);
#else
  *(uint4*)l = *(const uint4*)g;
#endif
}

// --- Kernel 1: block-wise 6-bit fake-quant of weights + pack to [pos][co][ci] bf16 ---
// Quant blocks = 32 consecutive flat (co,ci,kh,kw) elems == exactly one wave32.
__global__ __launch_bounds__(256) void dequant_pack(const float* __restrict__ w,
                                                    unsigned short* __restrict__ wp) {
  int tid = blockIdx.x * 256 + threadIdx.x;
  float v = w[tid];
  float amax = fabsf(v);
  #pragma unroll
  for (int off = 16; off > 0; off >>= 1)
    amax = fmaxf(amax, __shfl_xor(amax, off, 32));
  float scale = amax * (1.0f / 31.0f);
  if (scale == 0.0f) scale = 1.0f;
  float q = rintf(v / scale);
  q = fminf(fmaxf(q, -32.0f), 31.0f);
  float dq = q * scale;

  int o   = tid / KTAP;
  int r   = tid - o * KTAP;     // ci*9 + kh*3 + kw
  int ci  = r / 9;
  int pos = r - ci * 9;
  wp[(pos * COUT + o) * CIN + ci] = f2bf(dq);
}

// --- Kernel 2: x NCHW f32 -> zero-haloed NHWC bf16 [32][58][58][256] ---
// Halo removes ALL padding predication from the GEMM staging loads.
__global__ __launch_bounds__(256) void x_pad_nhwc_bf16(const float* __restrict__ x,
                                                       unsigned short* __restrict__ xq) {
  unsigned idx = blockIdx.x * 256 + threadIdx.x;   // ((n*58+hp)*58+wp)*256 + c
  int c  = idx & (CIN - 1);
  unsigned t = idx >> 8;
  int n  = t / (HP * HP);
  int r  = t - n * (HP * HP);
  int hp = r / HP;
  int wq = r - hp * HP;
  int ih = hp - 1, iw = wq - 1;
  unsigned short v = 0;
  if ((unsigned)ih < (unsigned)HH && (unsigned)iw < (unsigned)WWID)
    v = f2bf(x[(n * CIN + c) * HWX + ih * WWID + iw]);
  xq[idx] = v;
}

// --- Kernel 3: implicit-GEMM conv, bf16 WMMA, async double-buffered LDS pipeline ---
// Grid (784, 4); 256 threads = 8 waves in 2(M) x 4(N); wave tile 64x32 = 4x2 frags.
__global__ __launch_bounds__(256) void conv_wmma(const unsigned short* __restrict__ xq,
                                                 const unsigned short* __restrict__ wp,
                                                 const float* __restrict__ bias,
                                                 float* __restrict__ out) {
  // [A buf0][A buf1][B buf0][B buf1]; reused as 64x132 f32 transpose buffer in epilogue
  __shared__ __attribute__((aligned(16))) unsigned short AB[4 * ABUF];  // 40960 B

  const int tid   = threadIdx.x;
  const int lane  = tid & 31;
  const int wave  = tid >> 5;
  const int tileM = blockIdx.x;
  const int tileN = blockIdx.y;

  const int wm    = (wave >> 2) * 64;
  const int wn    = (wave & 3) * 32;
  const int l16   = lane & 15;
  const int grp   = lane >> 4;
  const int aKoff = grp * 8;          // A frag: K0-7 / K8-15 (+16 for second half)
  const int bKoff = grp * 8;          // B frag: K0-7 / K8-15 (+16 for second half)
  const int qcol  = (tid & 3) * 8;

  // Staging chunks: 512 x 16B per tile, 2 per thread: (row = tid>>2 + 64j, col = qcol)
  int aRow[2];
  const unsigned short* aBase[2];
  #pragma unroll
  for (int j = 0; j < 2; ++j) {
    int p  = (tid >> 2) + j * 64;
    int m  = tileM * BM + p;
    int n  = m / HWX;
    int hw = m - n * HWX;
    int oh = hw / WWID;
    int ow = hw - oh * WWID;
    aRow[j]  = p;
    aBase[j] = xq + ((n * HP + oh) * HP + ow) * CIN + qcol;  // tap adds (kh*58+kw)*256
  }
  const unsigned short* bBase0 = wp + (tileN * BN) * CIN + qcol;

  auto issue = [&](int s, int buf) {
    const int pos = s >> 3;
    const int kc  = (s & 7) << 5;
    const int kh  = pos / 3;
    const int kw  = pos - 3 * kh;
    const int aOff = (kh * HP + kw) * CIN + kc;
    const unsigned short* gb = bBase0 + pos * (COUT * CIN) + kc;
    unsigned short* As = &AB[buf * ABUF];
    unsigned short* Bs = &AB[(2 + buf) * ABUF];
    #pragma unroll
    for (int j = 0; j < 2; ++j) {
      cp16_g2l(aBase[j] + aOff,      &As[aRow[j] * AST + qcol]);
      cp16_g2l(gb + aRow[j] * CIN,   &Bs[aRow[j] * AST + qcol]);
    }
  };

  v8f acc[4][2] = {};

  issue(0, 0);
  issue(1, 1);

  for (int s = 0; s < NSTG; ++s) {
    const int buf = s & 1;
#if USE_ASYNC
    if (s < NSTG - 1) __builtin_amdgcn_s_wait_asynccnt(4);  // own stage-s loads done
    else              __builtin_amdgcn_s_wait_asynccnt(0);
#endif
    __syncthreads();                                        // everyone's stage-s data ready

    const unsigned short* As = &AB[buf * ABUF];
    const unsigned short* Bs = &AB[(2 + buf) * ABUF];
    union Frag { uint4 q[2]; v16bf v; };
    Frag a[4], b[2];
    #pragma unroll
    for (int mi = 0; mi < 4; ++mi) {
      const unsigned short* pA = &As[(wm + mi * 16 + l16) * AST + aKoff];
      a[mi].q[0] = *(const uint4*)(pA);        // K = aKoff .. aKoff+7
      a[mi].q[1] = *(const uint4*)(pA + 16);   // K = aKoff+16 .. aKoff+23
    }
    #pragma unroll
    for (int ni = 0; ni < 2; ++ni) {
      const unsigned short* pB = &Bs[(wn + ni * 16 + l16) * AST + bKoff];
      b[ni].q[0] = *(const uint4*)(pB);        // K = bKoff .. bKoff+7
      b[ni].q[1] = *(const uint4*)(pB + 16);   // K = bKoff+16 .. bKoff+23
    }
    #pragma unroll
    for (int mi = 0; mi < 4; ++mi)
      #pragma unroll
      for (int ni = 0; ni < 2; ++ni)
        acc[mi][ni] = __builtin_amdgcn_wmma_f32_16x16x32_bf16(
            false, a[mi].v, false, b[ni].v, (short)0, acc[mi][ni], false, false);

    __syncthreads();                           // everyone done reading this buffer
    if (s + 2 < NSTG) issue(s + 2, buf);       // refill it two stages ahead
  }

  // --- Epilogue: LDS transpose -> coalesced float4 stores (+bias), 2 halves of 64 co ---
  float* T = (float*)AB;                       // 64 x 132 f32 = 33792 B (padded vs banks)
  for (int half = 0; half < 2; ++half) {
    if (((wave >> 1) & 1) == half) {
      const int coBase = (wave & 1) * 32;
      #pragma unroll
      for (int ni = 0; ni < 2; ++ni)
        #pragma unroll
        for (int mi = 0; mi < 4; ++mi)
          #pragma unroll
          for (int i = 0; i < 8; ++i)
            T[(coBase + ni * 16 + l16) * 132 + (wm + mi * 16 + grp * 8 + i)] =
                acc[mi][ni][i];
    }
    __syncthreads();
    #pragma unroll
    for (int it = 0; it < 8; ++it) {
      int chunk = it * 256 + tid;              // 2048 float4 chunks = 64co x 128m
      int coL = chunk >> 5;
      int q   = chunk & 31;
      int co  = tileN * BN + half * 64 + coL;
      int m   = tileM * BM + q * 4;
      int n   = m / HWX;
      int hw  = m - n * HWX;
      float4 v = *(const float4*)&T[coL * 132 + q * 4];
      float bv = bias[co];
      v.x += bv; v.y += bv; v.z += bv; v.w += bv;
      *(float4*)&out[(n * COUT + co) * HWX + hw] = v;
    }
    __syncthreads();
  }
}

extern "C" void kernel_launch(void* const* d_in, const int* in_sizes, int n_in,
                              void* d_out, int out_size, void* d_ws, size_t ws_size,
                              hipStream_t stream) {
  const float* x    = (const float*)d_in[0];   // [32,256,56,56]
  const float* w    = (const float*)d_in[1];   // [512,256,3,3]
  const float* bias = (const float*)d_in[2];   // [512]
  float* out        = (float*)d_out;           // [32,512,56,56]

  unsigned short* wpk = (unsigned short*)d_ws;                                   // 2.36 MB
  unsigned short* xq  = (unsigned short*)((char*)d_ws + (size_t)WPACK_ELEMS * 2); // 55.1 MB

  dequant_pack<<<(COUT * KTAP) / 256, 256, 0, stream>>>(w, wpk);
  x_pad_nhwc_bf16<<<XPAD_ELEMS / 256, 256, 0, stream>>>(x, xq);

  dim3 grid(MTOT / BM, COUT / BN);
  conv_wmma<<<grid, 256, 0, stream>>>(xq, wpk, bias, out);
}
